// SwinWindowAttention_30657476559227
// MI455X (gfx1250) — matrix operand order repaired and checked by reading, not compiled, not run on hardware
//
#include <hip/hip_runtime.h>
#include <hip/hip_bf16.h>

typedef _Float16 v16h __attribute__((ext_vector_type(16)));
typedef _Float16 v8h  __attribute__((ext_vector_type(8)));
typedef float    v8f  __attribute__((ext_vector_type(8)));

// ---------------- LDS layout (bytes) ----------------
// sx   : f16 [64][136]        (window acts / attn-out staging)   17408 B
// sq   : f16 [8][64][24]      (q, per head, row=token)           24576 B
// sk   : f16 [8][64][24]      (k, per head, row=token)           24576 B
// sv   : f16 [8][16][72]      (v transposed: [head][ch][token])  18432 B
// satt : f32 [64][68] x4 waves (scores; f16 probs overlay rows)  69632 B
#define LDS_X    0
#define LDS_Q    17408
#define LDS_K    41984
#define LDS_V    66560
#define LDS_ATT  84992
#define LDS_TOTAL (84992 + 4*17408)   // 154624 bytes dynamic LDS

// ---------------- workspace layout (bytes) ----------------
// [0,131072)        biasExp f32 [8][64][64]
// [131072,229376)   qkv_w  f16, pre-swizzled B fragments (96 frags x 512)
// [229376,262144)   proj_w f16, pre-swizzled B fragments (32 frags x 512)

__global__ __launch_bounds__(256)
void swin_prep_kernel(const float* __restrict__ qkv_w,
                      const float* __restrict__ proj_w,
                      const float* __restrict__ rel_bias,
                      const int*   __restrict__ rel_index,
                      float*       __restrict__ biasExp,
                      _Float16*    __restrict__ wqkv,
                      _Float16*    __restrict__ wproj) {
  int idx = blockIdx.x * blockDim.x + threadIdx.x;
  if (idx < 32768) {
    // biasExp[h][t][s] = rel_bias[rel_index[t*64+s]][h]
    int h  = idx >> 12;
    int ts = idx & 4095;
    biasExp[idx] = rel_bias[rel_index[ts] * 8 + h];
  } else if (idx < 81920) {
    // qkv_w [128][384] -> B fragments for 16x16x32 f16 WMMA
    int e = idx - 32768;
    int f = e >> 9, r = e & 511;
    int l = r >> 4, i = r & 15;
    int kt = f / 24, nt = f - kt * 24;
    int kg = kt * 32 + ((l >> 4) << 3) + (i & 7) + ((i >> 3) << 4);   // K index (0..127)
    int ng = (nt << 4) + (l & 15);            // N index (0..383)
    wqkv[e] = (_Float16)qkv_w[kg * 384 + ng];
  } else if (idx < 98304) {
    int e = idx - 81920;
    int f = e >> 9, r = e & 511;
    int l = r >> 4, i = r & 15;
    int kt = f >> 3, nt = f & 7;
    int kg = kt * 32 + ((l >> 4) << 3) + (i & 7) + ((i >> 3) << 4);
    int ng = (nt << 4) + (l & 15);
    wproj[e] = (_Float16)proj_w[kg * 128 + ng];
  }
}

static __device__ __forceinline__ v16h frag16(const _Float16* p0, const _Float16* p1) {
  union { v16h w; v8h h[2]; } u;
  u.h[0] = *(const v8h*)p0;
  u.h[1] = *(const v8h*)p1;
  return u.w;
}

__global__ __launch_bounds__(128)
void swin_attn_kernel(const float* __restrict__ x,
                      const float* __restrict__ qkv_b,
                      const float* __restrict__ proj_b,
                      const _Float16* __restrict__ wqkv,
                      const _Float16* __restrict__ wproj,
                      const float* __restrict__ biasExp,
                      float* __restrict__ out) {
  extern __shared__ char smem[];
  _Float16* sx = (_Float16*)(smem + LDS_X);
  _Float16* sq = (_Float16*)(smem + LDS_Q);
  _Float16* sk = (_Float16*)(smem + LDS_K);
  _Float16* sv = (_Float16*)(smem + LDS_V);

  const int tid  = threadIdx.x;
  const int wave = tid >> 5;
  const int lane = tid & 31;
  const int hl   = lane >> 4;   // wave half (0/1)
  const int l15  = lane & 15;

  const int wid = blockIdx.x;            // 9216 windows = 4 * 48 * 48
  const int b   = wid / 2304;
  const int r0  = wid - b * 2304;
  const int wy  = r0 / 48;
  const int wx  = r0 - wy * 48;

  // ---------- Phase 1: gather rolled window -> LDS f16 ----------
  {
    const int p   = tid & 63;            // pixel in window
    const int ch0 = tid >> 6;            // 0/1 channel phase
    const int ty = p >> 3, tx = p & 7;
    const int sh = (wy * 8 + ty + 4) % 384;
    const int sw = (wx * 8 + tx + 4) % 384;
    const float* px = x + (size_t)b * 128 * 147456 + (size_t)sh * 384 + sw;
    for (int c = ch0; c < 128; c += 2)
      sx[p * 136 + c] = (_Float16)px[(size_t)c * 147456];
  }
  __syncthreads();

  // ---------- Phase 2: QKV = win @ qkv_w + b ----------
  for (int t = wave; t < 96; t += 4) {
    const int mt = t / 24, nt = t - mt * 24;
    const int row = l15 + mt * 16;
    v8f acc = {};
    for (int kt = 0; kt < 4; ++kt) {
      const int bh = hl * 8 + kt * 32;
      v16h a  = frag16(sx + row * 136 + bh, sx + row * 136 + bh + 16);
      const _Float16* pw = wqkv + (size_t)(kt * 24 + nt) * 512 + lane * 16;
      v16h bb = frag16(pw, pw + 8);
      acc = __builtin_amdgcn_wmma_f32_16x16x32_f16(false, a, false, bb,
                                                   (short)0, acc, false, false);
    }
    const float bias = qkv_b[nt * 16 + l15];
    const int sec  = nt >> 3;            // 0=q, 1=k, 2=v  (wave-uniform)
    const int hsec = nt & 7;             // head
    if (sec < 2) {
      _Float16* dst = (sec == 0 ? sq : sk)
                      + ((size_t)hsec * 64 + hl * 8 + mt * 16) * 24 + l15;
      #pragma unroll
      for (int v = 0; v < 8; ++v)
        dst[(size_t)v * 24] = (_Float16)(acc[v] + bias);
    } else {
      // transposed v: 8 consecutive tokens -> one b128 store
      _Float16* dst = sv + ((size_t)hsec * 16 + l15) * 72 + hl * 8 + mt * 16;
      v8h hv;
      #pragma unroll
      for (int v = 0; v < 8; ++v) hv[v] = (_Float16)(acc[v] + bias);
      *(v8h*)dst = hv;
    }
  }
  __syncthreads();

  // ---------- Phase 3: attention, 2 heads per wave ----------
  float* sa = (float*)(smem + LDS_ATT + wave * 17408);   // [64][68] f32
  const float scale = 0.25f;                              // 16^-0.5
  for (int hh = 0; hh < 2; ++hh) {
    const int h = wave * 2 + hh;

    // scores = q @ k^T  (head dim 16 padded to K=32 with zeros)
    for (int mt = 0; mt < 4; ++mt) {
      const int row = l15 + mt * 16;
      union { v16h w; v8h h2[2]; } ua;
      v8h z = {};
      ua.h2[0] = *(const v8h*)(sq + (h * 64 + row) * 24 + hl * 8);
      ua.h2[1] = z;                                       // K 16..31 = 0
      for (int nt = 0; nt < 4; ++nt) {
        union { v16h w; v8h h2[2]; } ub;
        ub.h2[1] = z;                                     // K 16..31 = 0
        {
          const _Float16* pk = sk + (h * 64 + (l15 + nt * 16)) * 24 + hl * 8;
          ub.h2[0] = *(const v8h*)pk;
        }
        v8f zc = {};
        v8f d = __builtin_amdgcn_wmma_f32_16x16x32_f16(false, ua.w, false, ub.w,
                                                       (short)0, zc, false, false);
        #pragma unroll
        for (int v = 0; v < 8; ++v)
          sa[(v + hl * 8 + mt * 16) * 68 + (l15 + nt * 16)] = d[v];
      }
    }

    // row-wise softmax(scale * s + bias); write f16 probs overlaying rows
    #pragma unroll
    for (int rr = 0; rr < 2; ++rr) {
      const int r = lane + rr * 32;
      const float4* prow = (const float4*)(sa + r * 68);
      const float4* pb4  = (const float4*)(biasExp + ((size_t)h * 64 + r) * 64);
      float vals[64];
      float mx = -1e30f;
      #pragma unroll
      for (int j4 = 0; j4 < 16; ++j4) {
        float4 s4 = prow[j4];
        float4 b4 = pb4[j4];
        float u0 = fmaf(s4.x, scale, b4.x);
        float u1 = fmaf(s4.y, scale, b4.y);
        float u2 = fmaf(s4.z, scale, b4.z);
        float u3 = fmaf(s4.w, scale, b4.w);
        vals[j4 * 4 + 0] = u0;
        vals[j4 * 4 + 1] = u1;
        vals[j4 * 4 + 2] = u2;
        vals[j4 * 4 + 3] = u3;
        mx = fmaxf(mx, fmaxf(fmaxf(u0, u1), fmaxf(u2, u3)));
      }
      float s = 0.f;
      #pragma unroll
      for (int j = 0; j < 64; ++j) {
        float e = __expf(vals[j] - mx);
        vals[j] = e;
        s += e;
      }
      const float inv = 1.0f / s;
      _Float16* pr = (_Float16*)((char*)sa + (size_t)r * 272);
      #pragma unroll
      for (int j8 = 0; j8 < 8; ++j8) {
        v8h hv;
        #pragma unroll
        for (int e = 0; e < 8; ++e) hv[e] = (_Float16)(vals[j8 * 8 + e] * inv);
        *(v8h*)(pr + j8 * 8) = hv;
      }
    }

    // out_head = P @ v   (K = 64, two ksteps)
    for (int mt = 0; mt < 4; ++mt) {
      const int row = l15 + mt * 16;
      const _Float16* prow = (const _Float16*)((const char*)sa + (size_t)row * 272);
      v8f acc = {};
      #pragma unroll
      for (int ks = 0; ks < 2; ++ks) {
        const int bh = hl * 8 + ks * 32;
        v16h a  = frag16(prow + bh, prow + bh + 16);
        const _Float16* pv = sv + (h * 16 + l15) * 72 + hl * 8 + ks * 32;
        v16h bb = frag16(pv, pv + 16);
        acc = __builtin_amdgcn_wmma_f32_16x16x32_f16(false, a, false, bb,
                                                     (short)0, acc, false, false);
      }
      #pragma unroll
      for (int v = 0; v < 8; ++v) {
        const int m = v + hl * 8 + mt * 16;
        sx[m * 136 + h * 16 + l15] = (_Float16)acc[v];   // reuse sx as hbuf
      }
    }
  }
  __syncthreads();

  // ---------- Phase 4: proj + scatter (inverse roll) ----------
  for (int t = wave; t < 32; t += 4) {
    const int mt = t >> 3, nt = t & 7;
    const int row = l15 + mt * 16;
    v8f acc = {};
    for (int kt = 0; kt < 4; ++kt) {
      const int bh = hl * 8 + kt * 32;
      v16h a  = frag16(sx + row * 136 + bh, sx + row * 136 + bh + 16);
      const _Float16* pw = wproj + (size_t)(kt * 8 + nt) * 512 + lane * 16;
      v16h bb = frag16(pw, pw + 8);
      acc = __builtin_amdgcn_wmma_f32_16x16x32_f16(false, a, false, bb,
                                                   (short)0, acc, false, false);
    }
    const int   n  = nt * 16 + l15;
    const float pb = proj_b[n];
    float* po = out + ((size_t)b * 128 + n) * 147456;

    // lane's 8 accumulator values = one window row (ty fixed), tx = 0..7
    const int mbase = hl * 8 + mt * 16;
    const int ty = mbase >> 3;
    const int oh = (wy * 8 + ty + 4) % 384;
    float* prw = po + (size_t)oh * 384;
    if (wx != 47) {
      const int ow0 = wx * 8 + 4;          // no wrap; 16B-aligned
      float4 lo, hi;
      lo.x = acc[0] + pb; lo.y = acc[1] + pb; lo.z = acc[2] + pb; lo.w = acc[3] + pb;
      hi.x = acc[4] + pb; hi.y = acc[5] + pb; hi.z = acc[6] + pb; hi.w = acc[7] + pb;
      *(float4*)(prw + ow0)     = lo;
      *(float4*)(prw + ow0 + 4) = hi;
    } else {
      #pragma unroll
      for (int v = 0; v < 8; ++v)
        prw[(wx * 8 + v + 4) % 384] = acc[v] + pb;
    }
  }
}

extern "C" void kernel_launch(void* const* d_in, const int* in_sizes, int n_in,
                              void* d_out, int out_size, void* d_ws, size_t ws_size,
                              hipStream_t stream) {
  const float* x         = (const float*)d_in[0];
  const float* qkv_w     = (const float*)d_in[1];
  const float* qkv_b     = (const float*)d_in[2];
  const float* proj_w    = (const float*)d_in[3];
  const float* proj_b    = (const float*)d_in[4];
  const float* rel_bias  = (const float*)d_in[5];
  const int*   rel_index = (const int*)d_in[6];

  float*    ws_bias  = (float*)d_ws;
  _Float16* ws_qkvw  = (_Float16*)((char*)d_ws + 131072);
  _Float16* ws_projw = (_Float16*)((char*)d_ws + 229376);

  swin_prep_kernel<<<384, 256, 0, stream>>>(qkv_w, proj_w, rel_bias, rel_index,
                                            ws_bias, ws_qkvw, ws_projw);

  swin_attn_kernel<<<9216, 128, LDS_TOTAL, stream>>>(x, qkv_b, proj_b,
                                                     ws_qkvw, ws_projw, ws_bias,
                                                     (float*)d_out);
}
